// EquivariantBlock_61701500174840
// MI455X (gfx1250) — matrix operand order repaired and checked by reading, not compiled, not run on hardware
//
#include <hip/hip_runtime.h>
#include <hip/hip_bf16.h>

// ---------------------------------------------------------------------------
// EGNN layer for MI455X (gfx1250, wave32, WMMA).
// Edge MLPs (166 GFLOP over 800k edges) run on v_wmma_f32_16x16x32_f16 with
// f32 accumulate; features staged f16 in LDS in the ISA A-fragment layout,
// weights pre-packed into B-fragment order for contiguous 128-bit loads.
// Scatter-add targets (h_agg: 25.6MB) are L2-resident on the 192MB L2, so
// f32 global atomics stay on-chip. HBM floor ~65us @ 23.3 TB/s.
// ---------------------------------------------------------------------------

typedef __attribute__((ext_vector_type(16))) _Float16 v16h;
typedef __attribute__((ext_vector_type(8)))  float    v8f;

#define N_NODES 50000
#define N_EDGES 800000
#define HDIM    128
#define EFEAT   16
#define MFEAT   276      // 2H + 1 + 3 + EF
#define KP1     288      // MFEAT padded to multiple of 32
#define KN1     256      // node MLP input width (2H)

union Frag16 { v16h v; float4 f4[2]; };

__device__ __forceinline__ float silu_f(float x)    { return x / (1.f + __expf(-x)); }
__device__ __forceinline__ float sigmoid_f(float x) { return 1.f / (1.f + __expf(-x)); }

// A-fragment (16x32 f16, ISA 7.12.2): lane m=l&15; lanes16-31 take K+8/K+24.
// Two 16-byte LDS reads per k-step.
__device__ __forceinline__ v16h lds_a_frag(const _Float16* p /*row base + kbase*/, int koff) {
  Frag16 u;
  u.f4[0] = *reinterpret_cast<const float4*>(p + koff);        // VGPRs 0-3: K = koff..koff+7
  u.f4[1] = *reinterpret_cast<const float4*>(p + 16 + koff);   // VGPRs 4-7: K = 16+koff..
  return u.v;
}

// B-fragment from weights pre-packed so each lane reads 8 consecutive dwords.
__device__ __forceinline__ v16h glb_b_frag(const unsigned int* __restrict__ pack,
                                           int kt, int nt, int lane) {
  Frag16 u;
  const float4* p = reinterpret_cast<const float4*>(pack + ((((kt * 8 + nt) * 32) + lane) << 3));
  u.f4[0] = p[0];
  u.f4[1] = p[1];
  return u.v;
}

__device__ __forceinline__ v8f wmma_f16(v16h a, v16h b, v8f c) {
  return __builtin_amdgcn_wmma_f32_16x16x32_f16(false, a, false, b, (short)0, c, false, false);
}

// ---------------------------------------------------------------------------
// Prep kernels (re-run every launch: deterministic, graph-capture safe)
// ---------------------------------------------------------------------------
__global__ void egnn_zero_kernel(float* __restrict__ p, int n) {
  for (int i = blockIdx.x * blockDim.x + threadIdx.x; i < n; i += gridDim.x * blockDim.x)
    p[i] = 0.f;
}

// Pack f32 weight [K x 128] into f16 WMMA-B fragment order, K padded to Kp.
// dword t = ((kt*8+nt)*32+lane)*8 + j holds (W[k][n], W[k+1][n]),
// k = kt*32 + (lane>>4)*8 + 2j + (j>=4 ? 8 : 0), n = nt*16 + (lane&15).
__global__ void egnn_pack_w_kernel(const float* __restrict__ src, int K, int Kp,
                                   unsigned int* __restrict__ dst) {
  const int t = blockIdx.x * blockDim.x + threadIdx.x;
  const int total = (Kp >> 5) * 2048;
  if (t >= total) return;
  const int j    = t & 7;
  const int lane = (t >> 3) & 31;
  const int nt   = (t >> 8) & 7;
  const int kt   = t >> 11;
  const int n = nt * 16 + (lane & 15);
  const int k = kt * 32 + (lane >> 4) * 8 + 2 * j + ((j >= 4) ? 8 : 0);
  const float lo = (k     < K) ? src[(size_t)k * HDIM + n]       : 0.f;
  const float hi = (k + 1 < K) ? src[(size_t)(k + 1) * HDIM + n] : 0.f;
  union { _Float16 h[2]; unsigned int u; } pk;
  pk.h[0] = (_Float16)lo;
  pk.h[1] = (_Float16)hi;
  dst[t] = pk.u;
}

// ---------------------------------------------------------------------------
// Edge kernel: 256 threads, 128 edges / block (8 waves x 16 edges)
// ---------------------------------------------------------------------------
__global__ __launch_bounds__(256)
void egnn_edge_kernel(const float* __restrict__ h,
                      const float* __restrict__ coords,
                      const float* __restrict__ a,
                      const int*   __restrict__ src,
                      const int*   __restrict__ dst,
                      const float* __restrict__ be1,
                      const float* __restrict__ be2,
                      const float* __restrict__ Wa,
                      const float* __restrict__ ba,
                      const float* __restrict__ bc1,
                      const float* __restrict__ bc2,
                      const float* __restrict__ Wc3,
                      const unsigned int* __restrict__ We1p,
                      const unsigned int* __restrict__ We2p,
                      const unsigned int* __restrict__ Wc1p,
                      const unsigned int* __restrict__ Wc2p,
                      float* __restrict__ h_agg,
                      float* __restrict__ x_agg) {
  extern __shared__ __align__(16) char smem[];
  _Float16* fbuf  = (_Float16*)smem;                       // 128*288*2 = 73728 B
  _Float16* h1e   = (_Float16*)(smem + 73728);             // 32768 B
  _Float16* h1c   = (_Float16*)(smem + 73728 + 32768);     // 32768 B
  float*    sdiff = (float*)(smem + 139264);               // 128*4*4 = 2048 B
  float*    sinv  = (float*)(smem + 141312);               // 512 B
  float*    attb  = (float*)(smem + 141824);               // 512 B
  float*    cmb   = (float*)(smem + 142336);               // 512 B
  _Float16* mhbuf = fbuf;  // overlay: fbuf is dead after layer-1 barrier

  const int tid  = threadIdx.x;
  const int eblk = blockIdx.x * 128;

  // ---- stage f = [h_src, h_dst, r, |dx|, a, pad] as f16 in LDS ----
  if (tid < 128) {
    const int e = eblk + tid;
    const int s = src[e], d = dst[e];
    const float dx = coords[3 * s + 0] - coords[3 * d + 0];
    const float dy = coords[3 * s + 1] - coords[3 * d + 1];
    const float dz = coords[3 * s + 2] - coords[3 * d + 2];
    const float r  = sqrtf(dx * dx + dy * dy + dz * dz);
    sdiff[4 * tid + 0] = dx; sdiff[4 * tid + 1] = dy; sdiff[4 * tid + 2] = dz;
    sinv[tid] = 1.f / (r + 1.f);
    _Float16* frow = fbuf + tid * KP1;
    frow[256] = (_Float16)r;
    frow[257] = (_Float16)fabsf(dx);
    frow[258] = (_Float16)fabsf(dy);
    frow[259] = (_Float16)fabsf(dz);
#pragma unroll
    for (int c = 0; c < EFEAT; ++c) frow[260 + c] = (_Float16)a[(size_t)e * EFEAT + c];
#pragma unroll
    for (int c = 276; c < 288; ++c) frow[c] = (_Float16)0.f;
  }
  {
    const int el   = tid >> 1;
    const int e    = eblk + el;
    const int node = (tid & 1) ? dst[e] : src[e];
    const float4* hp = reinterpret_cast<const float4*>(h + (size_t)node * HDIM);
    _Float16* frow = fbuf + el * KP1 + (tid & 1) * HDIM;
#pragma unroll 8
    for (int c = 0; c < 32; ++c) {
      const float4 v = hp[c];
      frow[4 * c + 0] = (_Float16)v.x; frow[4 * c + 1] = (_Float16)v.y;
      frow[4 * c + 2] = (_Float16)v.z; frow[4 * c + 3] = (_Float16)v.w;
    }
  }
  __syncthreads();

  const int wave = tid >> 5;
  const int lane = tid & 31;
  const int ncol = lane & 15;
  const int half = lane >> 4;
  const int koff = half * 8;

  // ---- layer 1 (both MLPs share A): [16 x 288] @ [288 x 128] ----
  v8f acce[8], accc[8];
  const v8f vzero = {0.f, 0.f, 0.f, 0.f, 0.f, 0.f, 0.f, 0.f};
#pragma unroll
  for (int nt = 0; nt < 8; ++nt) { acce[nt] = vzero; accc[nt] = vzero; }
  const _Float16* fr = fbuf + (wave * 16 + ncol) * KP1;
  for (int kt = 0; kt < 9; ++kt) {
    const v16h af = lds_a_frag(fr + kt * 32, koff);
#pragma unroll
    for (int nt = 0; nt < 8; ++nt) {
      acce[nt] = wmma_f16(af, glb_b_frag(We1p, kt, nt, lane), acce[nt]);
      accc[nt] = wmma_f16(af, glb_b_frag(Wc1p, kt, nt, lane), accc[nt]);
    }
  }
  __syncthreads();  // all waves done with fbuf; mhbuf may overlay it later

  // bias + SiLU -> hidden1 as f16 A-operand in LDS
  _Float16* h1e_w = h1e + wave * 16 * HDIM;
  _Float16* h1c_w = h1c + wave * 16 * HDIM;
#pragma unroll
  for (int nt = 0; nt < 8; ++nt) {
    const int n = nt * 16 + ncol;
    const float b_e = be1[n], b_c = bc1[n];
#pragma unroll
    for (int r = 0; r < 8; ++r) {
      const int m = r + 8 * half;
      h1e_w[m * HDIM + n] = (_Float16)silu_f(acce[nt][r] + b_e);
      h1c_w[m * HDIM + n] = (_Float16)silu_f(accc[nt][r] + b_c);
    }
  }

  // ---- layer 2: [16 x 128] @ [128 x 128] ----
  v8f a2e[8], a2c[8];
#pragma unroll
  for (int nt = 0; nt < 8; ++nt) { a2e[nt] = vzero; a2c[nt] = vzero; }
  const _Float16* h1e_r = h1e + (wave * 16 + ncol) * HDIM;
  const _Float16* h1c_r = h1c + (wave * 16 + ncol) * HDIM;
  for (int kt = 0; kt < 4; ++kt) {
    const v16h ae = lds_a_frag(h1e_r + kt * 32, koff);
    const v16h ac = lds_a_frag(h1c_r + kt * 32, koff);
#pragma unroll
    for (int nt = 0; nt < 8; ++nt) {
      a2e[nt] = wmma_f16(ae, glb_b_frag(We2p, kt, nt, lane), a2e[nt]);
      a2c[nt] = wmma_f16(ac, glb_b_frag(Wc2p, kt, nt, lane), a2c[nt]);
    }
  }

  // mh = silu(.)  -> LDS (f16);  att = sigmoid(mh.Wa+ba);  cm = silu(.)·Wc3
  _Float16* mh_w = mhbuf + wave * 16 * HDIM;
  float pa[8], pc[8];
#pragma unroll
  for (int r = 0; r < 8; ++r) { pa[r] = 0.f; pc[r] = 0.f; }
#pragma unroll
  for (int nt = 0; nt < 8; ++nt) {
    const int n = nt * 16 + ncol;
    const float b_e = be2[n], b_c = bc2[n];
    const float wav = Wa[n], w3 = Wc3[n];
#pragma unroll
    for (int r = 0; r < 8; ++r) {
      const int m = r + 8 * half;
      const float mhv = silu_f(a2e[nt][r] + b_e);
      mh_w[m * HDIM + n] = (_Float16)mhv;
      pa[r] += mhv * wav;
      pc[r] += silu_f(a2c[nt][r] + b_c) * w3;
    }
  }
#pragma unroll
  for (int mask = 1; mask < 16; mask <<= 1) {
#pragma unroll
    for (int r = 0; r < 8; ++r) {
      pa[r] += __shfl_xor(pa[r], mask, 32);
      pc[r] += __shfl_xor(pc[r], mask, 32);
    }
  }
  if (ncol == 0) {
    const float bav = ba[0];
#pragma unroll
    for (int r = 0; r < 8; ++r) {
      const int m = r + 8 * half;
      attb[wave * 16 + m] = sigmoid_f(pa[r] + bav);
      cmb[wave * 16 + m]  = pc[r];
    }
  }
  __syncthreads();

  // ---- scatter-add (L2-resident targets) ----
  if (tid < 128) {
    const int e = eblk + tid;
    const int d = dst[e];
    const float s = cmb[tid] * sinv[tid];
    atomicAdd(&x_agg[4 * d + 0], s * sdiff[4 * tid + 0]);
    atomicAdd(&x_agg[4 * d + 1], s * sdiff[4 * tid + 1]);
    atomicAdd(&x_agg[4 * d + 2], s * sdiff[4 * tid + 2]);
  }
  {
    const int el = tid >> 1;
    const int e  = eblk + el;
    const int d  = dst[e];
    const float attv = attb[el];
    const _Float16* mr = mhbuf + el * HDIM + (tid & 1) * 64;
    float* out = h_agg + (size_t)d * HDIM + (tid & 1) * 64;
#pragma unroll 8
    for (int c = 0; c < 64; ++c)
      atomicAdd(&out[c], attv * (float)mr[c]);
  }
}

// ---------------------------------------------------------------------------
// Node kernel: 256 threads, 128 nodes / block
// ---------------------------------------------------------------------------
__global__ __launch_bounds__(256)
void egnn_node_kernel(const float* __restrict__ h,
                      const float* __restrict__ coords,
                      const float* __restrict__ bn1,
                      const float* __restrict__ bn2,
                      const unsigned int* __restrict__ Wn1p,
                      const unsigned int* __restrict__ Wn2p,
                      const float* __restrict__ h_agg,
                      const float* __restrict__ x_agg,
                      float* __restrict__ h_out,
                      float* __restrict__ c_out) {
  extern __shared__ __align__(16) char smem[];
  _Float16* xbuf = (_Float16*)smem;               // 128*256*2 = 65536 B
  _Float16* h1   = (_Float16*)(smem + 65536);     // 32768 B

  const int tid  = threadIdx.x;
  const int nblk = blockIdx.x * 128;

  // stage [h | h_agg] as f16
  {
    const int nl = tid >> 1;
    int node = nblk + nl;
    if (node >= N_NODES) node = N_NODES - 1;  // clamp; OOB rows never stored
    const float* srcp = (tid & 1) ? (h_agg + (size_t)node * HDIM)
                                  : (h     + (size_t)node * HDIM);
    const float4* sp = reinterpret_cast<const float4*>(srcp);
    _Float16* xrow = xbuf + nl * KN1 + (tid & 1) * HDIM;
#pragma unroll 8
    for (int c = 0; c < 32; ++c) {
      const float4 v = sp[c];
      xrow[4 * c + 0] = (_Float16)v.x; xrow[4 * c + 1] = (_Float16)v.y;
      xrow[4 * c + 2] = (_Float16)v.z; xrow[4 * c + 3] = (_Float16)v.w;
    }
  }
  __syncthreads();

  const int wave = tid >> 5;
  const int lane = tid & 31;
  const int ncol = lane & 15;
  const int half = lane >> 4;
  const int koff = half * 8;
  const v8f vzero = {0.f, 0.f, 0.f, 0.f, 0.f, 0.f, 0.f, 0.f};

  // layer 1: [16 x 256] @ [256 x 128]
  v8f acc[8];
#pragma unroll
  for (int nt = 0; nt < 8; ++nt) acc[nt] = vzero;
  const _Float16* xr = xbuf + (wave * 16 + ncol) * KN1;
  for (int kt = 0; kt < 8; ++kt) {
    const v16h af = lds_a_frag(xr + kt * 32, koff);
#pragma unroll
    for (int nt = 0; nt < 8; ++nt)
      acc[nt] = wmma_f16(af, glb_b_frag(Wn1p, kt, nt, lane), acc[nt]);
  }
  _Float16* h1_w = h1 + wave * 16 * HDIM;
#pragma unroll
  for (int nt = 0; nt < 8; ++nt) {
    const int n = nt * 16 + ncol;
    const float b1 = bn1[n];
#pragma unroll
    for (int r = 0; r < 8; ++r) {
      const int m = r + 8 * half;
      h1_w[m * HDIM + n] = (_Float16)silu_f(acc[nt][r] + b1);
    }
  }

  // layer 2: [16 x 128] @ [128 x 128], residual add, store
  v8f a2[8];
#pragma unroll
  for (int nt = 0; nt < 8; ++nt) a2[nt] = vzero;
  const _Float16* h1_r = h1 + (wave * 16 + ncol) * HDIM;
  for (int kt = 0; kt < 4; ++kt) {
    const v16h af = lds_a_frag(h1_r + kt * 32, koff);
#pragma unroll
    for (int nt = 0; nt < 8; ++nt)
      a2[nt] = wmma_f16(af, glb_b_frag(Wn2p, kt, nt, lane), a2[nt]);
  }
#pragma unroll
  for (int nt = 0; nt < 8; ++nt) {
    const int n = nt * 16 + ncol;
    const float b2 = bn2[n];
#pragma unroll
    for (int r = 0; r < 8; ++r) {
      const int m = r + 8 * half;
      const int node = nblk + wave * 16 + m;
      if (node < N_NODES)
        h_out[(size_t)node * HDIM + n] = h[(size_t)node * HDIM + n] + a2[nt][r] + b2;
    }
  }

  if (tid < 128) {
    const int node = nblk + tid;
    if (node < N_NODES) {
      c_out[3 * node + 0] = coords[3 * node + 0] + x_agg[4 * node + 0];
      c_out[3 * node + 1] = coords[3 * node + 1] + x_agg[4 * node + 1];
      c_out[3 * node + 2] = coords[3 * node + 2] + x_agg[4 * node + 2];
    }
  }
}

// ---------------------------------------------------------------------------
extern "C" void kernel_launch(void* const* d_in, const int* in_sizes, int n_in,
                              void* d_out, int out_size, void* d_ws, size_t ws_size,
                              hipStream_t stream) {
  (void)in_sizes; (void)n_in; (void)out_size; (void)ws_size;
  const float* h      = (const float*)d_in[0];
  const float* coords = (const float*)d_in[1];
  const float* a      = (const float*)d_in[2];
  const int*   src    = (const int*)d_in[3];
  const int*   dst    = (const int*)d_in[4];
  const float* We1    = (const float*)d_in[5];
  const float* be1    = (const float*)d_in[6];
  const float* We2    = (const float*)d_in[7];
  const float* be2    = (const float*)d_in[8];
  const float* Wa     = (const float*)d_in[9];
  const float* ba     = (const float*)d_in[10];
  const float* Wn1    = (const float*)d_in[11];
  const float* bn1    = (const float*)d_in[12];
  const float* Wn2    = (const float*)d_in[13];
  const float* bn2    = (const float*)d_in[14];
  const float* Wc1    = (const float*)d_in[15];
  const float* bc1    = (const float*)d_in[16];
  const float* Wc2    = (const float*)d_in[17];
  const float* bc2    = (const float*)d_in[18];
  const float* Wc3    = (const float*)d_in[19];

  // workspace layout (all 16B aligned)
  char* ws = (char*)d_ws;
  float*        h_agg = (float*)ws;                      // 50000*128*4 = 25,600,000
  float*        x_agg = (float*)(ws + 25600000);         // 50000*4*4  =    800,000
  unsigned int* We1p  = (unsigned int*)(ws + 26400000);  // 288*128/2 dwords = 73,728 B
  unsigned int* Wc1p  = (unsigned int*)(ws + 26473728);
  unsigned int* We2p  = (unsigned int*)(ws + 26547456);  // 32,768 B
  unsigned int* Wc2p  = (unsigned int*)(ws + 26580224);
  unsigned int* Wn1p  = (unsigned int*)(ws + 26612992);  // 65,536 B
  unsigned int* Wn2p  = (unsigned int*)(ws + 26678528);

  float* h_out = (float*)d_out;
  float* c_out = h_out + (size_t)N_NODES * HDIM;

  // zero scatter targets (h_agg and x_agg are contiguous)
  egnn_zero_kernel<<<1024, 256, 0, stream>>>(h_agg, N_NODES * HDIM + N_NODES * 4);

  // pack weights into WMMA-B fragment order (f16)
  egnn_pack_w_kernel<<<72, 256, 0, stream>>>(We1, MFEAT, KP1, We1p);
  egnn_pack_w_kernel<<<72, 256, 0, stream>>>(Wc1, MFEAT, KP1, Wc1p);
  egnn_pack_w_kernel<<<32, 256, 0, stream>>>(We2, HDIM, HDIM, We2p);
  egnn_pack_w_kernel<<<32, 256, 0, stream>>>(Wc2, HDIM, HDIM, Wc2p);
  egnn_pack_w_kernel<<<64, 256, 0, stream>>>(Wn1, KN1, KN1, Wn1p);
  egnn_pack_w_kernel<<<32, 256, 0, stream>>>(Wn2, HDIM, HDIM, Wn2p);

  // edge phase: 800000 / 128 = 6250 blocks, 142,848 B dynamic LDS
  egnn_edge_kernel<<<N_EDGES / 128, 256, 142848, stream>>>(
      h, coords, a, src, dst, be1, be2, Wa, ba, bc1, bc2, Wc3,
      We1p, We2p, Wc1p, Wc2p, h_agg, x_agg);

  // node phase: ceil(50000/128) = 391 blocks, 98,304 B dynamic LDS
  egnn_node_kernel<<<(N_NODES + 127) / 128, 256, 98304, stream>>>(
      h, coords, bn1, bn2, Wn1p, Wn2p, h_agg, x_agg, h_out, c_out);
}
